// GCNNet_19731079758141
// MI455X (gfx1250) — matrix operand order repaired and checked by reading, not compiled, not run on hardware
//
#include <hip/hip_runtime.h>

// ---------------------------------------------------------------------------
// Types for CDNA5 WMMA
// ---------------------------------------------------------------------------
typedef __attribute__((ext_vector_type(16))) __bf16          v16bf;
typedef __attribute__((ext_vector_type(8)))  unsigned short  v8u;
typedef __attribute__((ext_vector_type(8)))  float           v8f;

#define N_NODES  50000
#define N_EDGES  800000
#define N_GRAPHS 512

// fp32 -> bf16 bits, round-to-nearest-even, pure integer ops
static __device__ __forceinline__ unsigned f2bfbits(float f) {
    unsigned u = __float_as_uint(f);
    return (u + 0x7FFFu + ((u >> 16) & 1u)) >> 16;
}

struct BfPair { v8u lo, hi; };   // 32 bytes = one 16-elem bf16 fragment

// ---------------------------------------------------------------------------
// GEMM: C[M,N] = A[M,K] @ B[K,N] (+bias)(+relu).
// A: bf16 row-major [M][K].  BT: bf16 TRANSPOSED weights [N][K].
// Block: 128 threads = 4 waves; tile 64x64; wave w -> rows [16w,16w+16).
// Fragments load straight from global as 2x b128 each (no LDS, no barriers):
//   A: lane = khalf*16 + m%16 ; a[e] = A[m][k0 + 16*(e/8) + 8*khalf + e%8]
//   B: lane = khalf*16 + n    ; b[e] = BT[n][k0 + 16*(e/8) + 8*khalf + e%8]
//   D: lane n = lane%16       ; d[v] = C[v + 8*(lane/16)][n]
// ---------------------------------------------------------------------------
#define TM 64
#define TN 64
#define TK 32

__global__ __launch_bounds__(128)
void gemm_wmma_bf16t(const unsigned short* __restrict__ A,
                     const unsigned short* __restrict__ BT,
                     const float* __restrict__ bias, float* __restrict__ C,
                     int M, int N, int K, int relu)
{
    const int tid   = threadIdx.x;
    const int lane  = tid & 31;
    const int wave  = tid >> 5;
    const int m0    = blockIdx.y * TM;
    const int n0    = blockIdx.x * TN;
    const int l16   = lane & 15;
    const int khalf = lane >> 4;

    // A-fragment row for this lane; clamp OOB rows (keeps EXEC all-ones for
    // WMMA; clamped rows produce garbage that is never stored).
    int mfrag = m0 + wave * 16 + l16;
    if (mfrag >= M) mfrag = M - 1;
    const unsigned short* arow = A + (size_t)mfrag * K + 8 * khalf;

    // B-fragment column base per sub-tile
    const unsigned short* bcol[4];
    #pragma unroll
    for (int nt = 0; nt < 4; ++nt)
        bcol[nt] = BT + (size_t)(n0 + nt * 16 + l16) * K + 8 * khalf;

    v8f acc[4] = {};

    #pragma unroll 4
    for (int k0 = 0; k0 < K; k0 += TK) {
        BfPair ap;
        ap.lo = *reinterpret_cast<const v8u*>(arow + k0);
        ap.hi = *reinterpret_cast<const v8u*>(arow + k0 + 16);
        v16bf a = __builtin_bit_cast(v16bf, ap);

        if (k0 + 4 * TK < K)   // keep the gfx1250 prefetch path warm
            __builtin_prefetch(arow + k0 + 4 * TK, 0, 3);

        #pragma unroll
        for (int nt = 0; nt < 4; ++nt) {
            BfPair bp;
            bp.lo = *reinterpret_cast<const v8u*>(bcol[nt] + k0);
            bp.hi = *reinterpret_cast<const v8u*>(bcol[nt] + k0 + 16);
            v16bf b = __builtin_bit_cast(v16bf, bp);
            acc[nt] = __builtin_amdgcn_wmma_f32_16x16x32_bf16(
                false, a, false, b, (short)0, acc[nt], false, false);
        }
    }

    // ---- epilogue
    #pragma unroll
    for (int nt = 0; nt < 4; ++nt) {
        #pragma unroll
        for (int v = 0; v < 8; ++v) {
            int m = m0 + wave * 16 + v + (khalf << 3);
            int n = n0 + nt * 16 + l16;
            if (m < M) {
                float r = acc[nt][v];
                if (bias) r += bias[n];
                if (relu) r = fmaxf(r, 0.0f);
                C[(size_t)m * N + n] = r;
            }
        }
    }
}

// ---------------------------------------------------------------------------
// fp32 -> bf16 convert (n multiple of 4): float4 read, 8B packed write
// ---------------------------------------------------------------------------
__global__ void cvt_bf16_kernel(const float* __restrict__ in,
                                unsigned short* __restrict__ out, int n)
{
    int t = blockIdx.x * blockDim.x + threadIdx.x;
    int base = t * 4;
    if (base < n) {
        float4 v = *reinterpret_cast<const float4*>(in + base);
        uint2 p;
        p.x = f2bfbits(v.x) | (f2bfbits(v.y) << 16);
        p.y = f2bfbits(v.z) | (f2bfbits(v.w) << 16);
        *reinterpret_cast<uint2*>(out + base) = p;
    }
}

// ---------------------------------------------------------------------------
// Weight convert + transpose: W[K][N] fp32 -> WT[N][K] bf16 (small matrices)
// ---------------------------------------------------------------------------
__global__ void transpose_bf16_kernel(const float* __restrict__ W,
                                      unsigned short* __restrict__ WT,
                                      int K, int N)
{
    int t = blockIdx.x * blockDim.x + threadIdx.x;
    if (t < K * N) {
        int k = t / N;
        int n = t - k * N;
        WT[(size_t)n * K + k] = (unsigned short)f2bfbits(W[t]);
    }
}

// ---------------------------------------------------------------------------
// Degree accumulation: deg[dst] += ew   (self-loop +1 added in dinv kernel)
// ---------------------------------------------------------------------------
__global__ void deg_kernel(const long long* __restrict__ ei,
                           const float* __restrict__ ew,
                           float* __restrict__ deg, int E)
{
    int e = blockIdx.x * blockDim.x + threadIdx.x;
    if (e < E) atomicAdd(&deg[(int)ei[(size_t)E + e]], ew[e]);
}

__global__ void dinv_kernel(float* __restrict__ deg, int n)
{
    int i = blockIdx.x * blockDim.x + threadIdx.x;
    if (i < n) {
        float d = deg[i] + 1.0f;            // self-loop weight 1
        deg[i] = (d > 0.0f) ? rsqrtf(d) : 0.0f;
    }
}

// ---------------------------------------------------------------------------
// Edge scatter: acc[dst,:] += h[src,:] * dinv[src]*ew*dinv[dst]
// One block per edge; coalesced row read, L2-resident atomics.
// ---------------------------------------------------------------------------
__global__ __launch_bounds__(256)
void scatter_kernel(const float* __restrict__ h,
                    const long long* __restrict__ ei,
                    const float* __restrict__ ew,
                    const float* __restrict__ dinv,
                    float* __restrict__ acc, int E, int F)
{
    __shared__ int   s_sh, d_sh;
    __shared__ float c_sh;
    int e = blockIdx.x;
    if (threadIdx.x == 0) {
        int s = (int)ei[e];
        int d = (int)ei[(size_t)E + e];
        s_sh = s; d_sh = d;
        c_sh = dinv[s] * ew[e] * dinv[d];
    }
    __syncthreads();
    const int s = s_sh, d = d_sh;
    const float c = c_sh;
    for (int f = threadIdx.x; f < F; f += blockDim.x)
        atomicAdd(&acc[(size_t)d * F + f], h[(size_t)s * F + f] * c);
}

// ---------------------------------------------------------------------------
// Finalize (in-place on acc): out = relu(acc + h*dinv^2 + bias), fshift=log2F
// ---------------------------------------------------------------------------
__global__ void finalize_kernel(const float* __restrict__ h,
                                const float* __restrict__ bias,
                                const float* __restrict__ dinv,
                                float* __restrict__ acc, int n, int fshift)
{
    int t = blockIdx.x * blockDim.x + threadIdx.x;
    if (t < (n << fshift)) {
        int i = t >> fshift;
        int f = t & ((1 << fshift) - 1);
        float di = dinv[i];
        float v = acc[t] + h[t] * di * di + bias[f];
        acc[t] = fmaxf(v, 0.0f);
    }
}

// ---------------------------------------------------------------------------
// Global max pool per graph (post-ReLU inputs >= 0, buffer zeroed ->
// uint-bit atomicMax == float max).
// ---------------------------------------------------------------------------
__global__ __launch_bounds__(256)
void pool_kernel(const float* __restrict__ h,
                 const long long* __restrict__ batch,
                 float* __restrict__ g, int n, int F)
{
    int i = blockIdx.x;
    int gid = (int)batch[i];
    for (int f = threadIdx.x; f < F; f += blockDim.x)
        atomicMax((unsigned*)&g[(size_t)gid * F + f],
                  __float_as_uint(h[(size_t)i * F + f]));
}

// ---------------------------------------------------------------------------
// Launch
// ---------------------------------------------------------------------------
extern "C" void kernel_launch(void* const* d_in, const int* in_sizes, int n_in,
                              void* d_out, int out_size, void* d_ws, size_t ws_size,
                              hipStream_t stream)
{
    const float*     x    = (const float*)d_in[0];
    const float*     ew   = (const float*)d_in[1];
    const float*     W1   = (const float*)d_in[2];
    const float*     b1   = (const float*)d_in[3];
    const float*     W2   = (const float*)d_in[4];
    const float*     b2   = (const float*)d_in[5];
    const float*     W3   = (const float*)d_in[6];
    const float*     b3   = (const float*)d_in[7];
    const float*     Wf1  = (const float*)d_in[8];
    const float*     bf1  = (const float*)d_in[9];
    const float*     Wf2  = (const float*)d_in[10];
    const float*     bf2  = (const float*)d_in[11];
    const long long* ei   = (const long long*)d_in[12];
    const long long* bat  = (const long long*)d_in[13];
    float*           out  = (float*)d_out;

    const int N = N_NODES, E = N_EDGES, G = N_GRAPHS;

    // workspace carve-out (256B aligned)
    char* ws = (char*)d_ws;
    size_t off = 0;
    auto carve = [&](size_t bytes) {
        void* p = ws + off;
        off += (bytes + 255) & ~(size_t)255;
        return p;
    };
    float*          dinv = (float*)carve((size_t)N * 4);              // 0.2 MB
    float*          bufH = (float*)carve((size_t)N * 512 * 4);        // 102 MB
    float*          bufA = (float*)carve((size_t)N * 512 * 4);        // 102 MB
    float*          bufB = (float*)carve((size_t)N * 256 * 4);        // 51 MB
    float*          bufP = (float*)carve((size_t)G * 512 * 4);        // 1 MB
    float*          bufF = (float*)carve((size_t)G * 1024 * 4);       // 2 MB
    unsigned short* bufXb = (unsigned short*)carve((size_t)N * 512 * 2); // 51 MB
    unsigned short* bufWt = (unsigned short*)carve((size_t)1024 * 1024 * 2); // 2 MB

    // ---- degree / dinv (shared by all 3 layers)
    hipMemsetAsync(dinv, 0, (size_t)N * 4, stream);
    deg_kernel<<<(E + 255) / 256, 256, 0, stream>>>(ei, ew, dinv, E);
    dinv_kernel<<<(N + 255) / 256, 256, 0, stream>>>(dinv, N);

    const int MB = (N + TM - 1) / TM;   // 782 row-blocks of nodes
    auto cvt = [&](const float* src, unsigned short* dst, int n) {
        cvt_bf16_kernel<<<(n / 4 + 255) / 256, 256, 0, stream>>>(src, dst, n);
    };
    auto tw = [&](const float* W, int K, int Nn) {
        transpose_bf16_kernel<<<(K * Nn + 255) / 256, 256, 0, stream>>>(W, bufWt, K, Nn);
    };

    // ---- GCN layer 1: x(128) -> 128
    cvt(x, bufXb, N * 128);
    tw(W1, 128, 128);
    gemm_wmma_bf16t<<<dim3(128 / TN, MB), 128, 0, stream>>>(bufXb, bufWt, nullptr,
                                                            bufH, N, 128, 128, 0);
    hipMemsetAsync(bufA, 0, (size_t)N * 128 * 4, stream);
    scatter_kernel<<<E, 128, 0, stream>>>(bufH, ei, ew, dinv, bufA, E, 128);
    finalize_kernel<<<(N * 128 + 255) / 256, 256, 0, stream>>>(bufH, b1, dinv,
                                                               bufA, N, 7);

    // ---- GCN layer 2: 128 -> 256
    cvt(bufA, bufXb, N * 128);
    tw(W2, 128, 256);
    gemm_wmma_bf16t<<<dim3(256 / TN, MB), 128, 0, stream>>>(bufXb, bufWt, nullptr,
                                                            bufH, N, 256, 128, 0);
    hipMemsetAsync(bufB, 0, (size_t)N * 256 * 4, stream);
    scatter_kernel<<<E, 256, 0, stream>>>(bufH, ei, ew, dinv, bufB, E, 256);
    finalize_kernel<<<(N * 256 + 255) / 256, 256, 0, stream>>>(bufH, b2, dinv,
                                                               bufB, N, 8);

    // ---- GCN layer 3: 256 -> 512
    cvt(bufB, bufXb, N * 256);
    tw(W3, 256, 512);
    gemm_wmma_bf16t<<<dim3(512 / TN, MB), 128, 0, stream>>>(bufXb, bufWt, nullptr,
                                                            bufH, N, 512, 256, 0);
    hipMemsetAsync(bufA, 0, (size_t)N * 512 * 4, stream);
    scatter_kernel<<<E, 256, 0, stream>>>(bufH, ei, ew, dinv, bufA, E, 512);
    finalize_kernel<<<(N * 512 + 255) / 256, 256, 0, stream>>>(bufH, b3, dinv,
                                                               bufA, N, 9);

    // ---- global max pool -> [512, 512]
    hipMemsetAsync(bufP, 0, (size_t)G * 512 * 4, stream);
    pool_kernel<<<N, 256, 0, stream>>>(bufA, bat, bufP, N, 512);

    // ---- FC1: [512,512]@[512,1024] + bias, relu
    cvt(bufP, bufXb, G * 512);
    tw(Wf1, 512, 1024);
    gemm_wmma_bf16t<<<dim3(1024 / TN, G / TM), 128, 0, stream>>>(bufXb, bufWt, bf1,
                                                                 bufF, G, 1024, 512, 1);
    // ---- FC2: [512,1024]@[1024,128] + bias -> d_out
    cvt(bufF, bufXb, G * 1024);
    tw(Wf2, 1024, 128);
    gemm_wmma_bf16t<<<dim3(128 / TN, G / TM), 128, 0, stream>>>(bufXb, bufWt, bf2,
                                                                out, G, 128, 1024, 0);
}
